// ViewLearner_86715389706491
// MI455X (gfx1250) — matrix operand order repaired and checked by reading, not compiled, not run on hardware
//
#include <hip/hip_runtime.h>

// ---------------- CDNA5 vector types ----------------
typedef __attribute__((ext_vector_type(16))) __bf16       v16bf;
typedef __attribute__((ext_vector_type(8)))  __bf16       v8bf;
typedef __attribute__((ext_vector_type(8)))  float        v8f;
typedef __attribute__((ext_vector_type(4)))  float        v4f;
typedef __attribute__((ext_vector_type(4)))  unsigned int v4u;

#define LDS_B_BYTES     32768                     // 256x64 bf16 W1 in WMMA B-frag layout
#define ROW_BYTES       528                       // 256 bf16 (512B) + 16B pad (bank spread)
#define A_TILE_BYTES    (16 * ROW_BYTES)          // 8448 per wave
#define WAVES_PER_BLOCK 8
#define BLOCK_THREADS   256
#define SMEM_BYTES      (LDS_B_BYTES + WAVES_PER_BLOCK * A_TILE_BYTES)  // 100352 < 320KB WGP LDS

// RNE f32 -> bf16 (bit trick; avoids header-struct dependence)
__device__ __forceinline__ unsigned short f32_to_bf16_rne(float f) {
    unsigned int u = __float_as_uint(f);
    unsigned int r = (u + 0x7FFFu + ((u >> 16) & 1u)) >> 16;
    return (unsigned short)r;
}

// -------- prep: W1[256][64] f32 (row-major, k-major) -> bf16 B-fragments --------
// B layout used here for V_WMMA_F32_16X16X32_BF16 (32x16, K x N), contiguous k order (A matches):
//   lane n (0..15): column N=n, VGPR j holds K = 2j, 2j+1  (K=0..15)
//   lane n+16:      column N=n, K = 16..31
// Fragment for (kb, t): 32 lanes x 16 bf16 (32B/lane), stored linearly.
__global__ void prep_w1_frags(const float* __restrict__ W1, unsigned short* __restrict__ Bfrag) {
    int o = blockIdx.x * blockDim.x + threadIdx.x;
    if (o >= 256 * 64) return;
    int i    =  o        & 15;   // element within lane fragment (K offset within half)
    int lane = (o >> 4)  & 31;
    int t    = (o >> 9)  & 3;    // N-tile (0..3)
    int kb   =  o >> 11;         // K-block (0..7)
    int n = t * 16 + (lane & 15);
    int k = kb * 32 + ((lane >> 4) << 4) + i;   // lanes 0-15: K 0..15 ; lanes 16-31: K 16..31
    Bfrag[o] = f32_to_bf16_rne(W1[k * 64 + n]);
}

// -------- fused gather + LayerNorm + MLP(WMMA) + ReLU + reduce --------
__global__ void __launch_bounds__(BLOCK_THREADS)
viewlearner_fused(const float* __restrict__ node_feat,
                  const float* __restrict__ edge_feat,
                  const float* __restrict__ gamma,
                  const float* __restrict__ beta,
                  const float* __restrict__ b1,
                  const float* __restrict__ W2,
                  const float* __restrict__ b2,
                  const int*   __restrict__ node_idx,
                  const int*   __restrict__ edge_idx,
                  const unsigned short* __restrict__ Bfrag,
                  float* __restrict__ out,
                  int E, int numTiles, int totalWaves)
{
    extern __shared__ char smem[];

    // ---- stage W1 B-fragments (32KB) into LDS, once per block ----
    {
        v4u* dst = (v4u*)smem;
        const v4u* src = (const v4u*)Bfrag;
        for (int i = threadIdx.x; i < LDS_B_BYTES / 16; i += BLOCK_THREADS)
            dst[i] = src[i];
    }
    __syncthreads();   // only block-wide barrier; tile loop below is wave-private

    const int lane = threadIdx.x & 31;
    const int wave = threadIdx.x >> 5;
    const int half = lane >> 4;       // 0 -> edge half (cols 0..127), 1 -> node half
    const int nloc = lane & 15;
    char* smemA = smem + LDS_B_BYTES + wave * A_TILE_BYTES;
    const int gwave = blockIdx.x * WAVES_PER_BLOCK + wave;

    // per-lane constants: gamma/beta for my 8 positions, b1/W2 for my 4 columns
    float g[8], be[8];
    {
        const float* gp = gamma + lane * 8;
        const float* bp = beta  + lane * 8;
        #pragma unroll
        for (int i = 0; i < 8; ++i) { g[i] = gp[i]; be[i] = bp[i]; }
    }
    float b1v[4], w2v[4];
    #pragma unroll
    for (int t = 0; t < 4; ++t) {
        b1v[t] = b1[t * 16 + nloc];
        w2v[t] = W2[t * 16 + nloc];
    }
    const float b2s = b2[0];

    for (int tile = gwave; tile < numTiles; tile += totalWaves) {
        const int base = tile * 16;

        // prefetch next tile's index cachelines (global_prefetch_b8)
        {
            int nb = base + totalWaves * 16;
            if (lane == 0 && nb < E) {
                __builtin_prefetch(edge_idx + nb, 0, 0);
                __builtin_prefetch(node_idx + nb, 0, 0);
            }
        }

        // preload 16 edge indices (lanes 0-15) + 16 node indices (lanes 16-31)
        int myE = base + nloc; if (myE >= E) myE = E - 1;
        int gidx = (half == 0) ? edge_idx[myE] : node_idx[myE];

        // ---- gather + LayerNorm -> bf16 rows in LDS (one row per iteration) ----
        for (int e = 0; e < 16; ++e) {
            int ridx = __shfl(gidx, half * 16 + e, 32);   // edge row for half 0, node row for half 1
            const float* src = (half == 0)
                ? edge_feat + (long)ridx * 128 + nloc * 8
                : node_feat + (long)ridx * 128 + nloc * 8;
            v4f x0 = *(const v4f*)src;
            v4f x1 = *(const v4f*)(src + 4);

            float xs[8] = {x0.x, x0.y, x0.z, x0.w, x1.x, x1.y, x1.z, x1.w};
            float s = 0.f, s2 = 0.f;
            #pragma unroll
            for (int i = 0; i < 8; ++i) { s += xs[i]; s2 += xs[i] * xs[i]; }
            #pragma unroll
            for (int off = 16; off >= 1; off >>= 1) {
                s  += __shfl_xor(s,  off, 32);
                s2 += __shfl_xor(s2, off, 32);
            }
            float mu  = s * (1.0f / 256.0f);
            float var = s2 * (1.0f / 256.0f) - mu * mu;
            float rs  = rsqrtf(var + 1e-5f);

            v8bf y;
            #pragma unroll
            for (int i = 0; i < 8; ++i)
                y[i] = (__bf16)((xs[i] - mu) * rs * g[i] + be[i]);
            *(v8bf*)(smemA + e * ROW_BYTES + lane * 16) = y;   // ds_store_b128
        }
        // wave-private LDS RAW: DS ops are in-order / compiler inserts s_wait_dscnt

        // ---- WMMA: [16x256] x [256x64] via 8 K-blocks x 4 N-tiles ----
        v8f acc[4] = {v8f{}, v8f{}, v8f{}, v8f{}};
        #pragma unroll
        for (int kb = 0; kb < 8; ++kb) {
            // A 16x32 bf16 frag (contiguous layout, matches B frags):
            // lanes 0-15 row M=lane: K 0..15
            // lanes 16-31 row M=lane-16: K 16..31
            union { v16bf v; v4u u[2]; } A;
            const char* ap = smemA + nloc * ROW_BYTES + kb * 64 + half * 32;
            A.u[0] = *(const v4u*)ap;          // ds_load_b128
            A.u[1] = *(const v4u*)(ap + 16);   // ds_load_b128
            #pragma unroll
            for (int t = 0; t < 4; ++t) {
                union { v16bf v; v4u u[2]; } B;
                const char* bp = smem + ((kb * 4 + t) * 32 + lane) * 32;
                B.u[0] = *(const v4u*)bp;
                B.u[1] = *(const v4u*)(bp + 16);
                acc[t] = __builtin_amdgcn_wmma_f32_16x16x32_bf16(
                    false, A.v, false, B.v, (short)0, acc[t], false, false);
            }
        }

        // ---- epilogue: h = relu(acc + b1); out = h . W2 + b2 ----
        // C layout: lane l holds (m = j + 8*half, n = t*16 + nloc) in acc[t][j]
        #pragma unroll
        for (int j = 0; j < 8; ++j) {
            float sacc = 0.f;
            #pragma unroll
            for (int t = 0; t < 4; ++t)
                sacc += fmaxf(acc[t][j] + b1v[t], 0.f) * w2v[t];
            sacc += __shfl_xor(sacc, 8, 32);   // reduce over the 16-lane half (n dimension)
            sacc += __shfl_xor(sacc, 4, 32);
            sacc += __shfl_xor(sacc, 2, 32);
            sacc += __shfl_xor(sacc, 1, 32);
            if (nloc == 0) {
                int eidx = base + j + half * 8;
                if (eidx < E) out[eidx] = sacc + b2s;
            }
        }
    }
}

extern "C" void kernel_launch(void* const* d_in, const int* in_sizes, int n_in,
                              void* d_out, int out_size, void* d_ws, size_t ws_size,
                              hipStream_t stream) {
    const float* node_feat = (const float*)d_in[0];
    const float* edge_feat = (const float*)d_in[1];
    const float* gamma     = (const float*)d_in[2];
    const float* beta      = (const float*)d_in[3];
    const float* W1        = (const float*)d_in[4];
    const float* b1        = (const float*)d_in[5];
    const float* W2        = (const float*)d_in[6];
    const float* b2        = (const float*)d_in[7];
    const int*   node_idx  = (const int*)d_in[8];
    const int*   edge_idx  = (const int*)d_in[9];
    float* out = (float*)d_out;

    int E = in_sizes[8];
    if (E <= 0) return;
    unsigned short* Bfrag = (unsigned short*)d_ws;   // 32KB of ws

    prep_w1_frags<<<(256 * 64 + 255) / 256, 256, 0, stream>>>(W1, Bfrag);

    int numTiles = (E + 15) / 16;
    int grid = 1024;                                  // 8192 wave32s; grid-stride over tiles
    if (grid > (numTiles + WAVES_PER_BLOCK - 1) / WAVES_PER_BLOCK)
        grid = (numTiles + WAVES_PER_BLOCK - 1) / WAVES_PER_BLOCK;
    int totalWaves = grid * WAVES_PER_BLOCK;

    viewlearner_fused<<<grid, BLOCK_THREADS, SMEM_BYTES, stream>>>(
        node_feat, edge_feat, gamma, beta, b1, W2, b2,
        node_idx, edge_idx, Bfrag, out, E, numTiles, totalWaves);
}
